// LilaModel_86698209837718
// MI455X (gfx1250) — matrix operand order repaired and checked by reading, not compiled, not run on hardware
//
#include <hip/hip_runtime.h>

typedef __attribute__((ext_vector_type(16))) _Float16     v16h;
typedef __attribute__((ext_vector_type(4)))  _Float16     v4h;
typedef __attribute__((ext_vector_type(8)))  float        v8f;
typedef __attribute__((ext_vector_type(8)))  unsigned int v8u;

#define WMMA(a, b, c) \
  __builtin_amdgcn_wmma_f32_16x16x32_f16(false, (a), false, (b), (short)0, (c), false, false)

// wave-private LDS ordering: DS ops of one wave complete in order; this only
// pins compiler ordering + inserts the dscnt wait (no barrier instruction).
#define WAVE_FENCE() __builtin_amdgcn_fence(__ATOMIC_ACQ_REL, "wavefront")

__device__ __forceinline__ float crelu1(float x) { return fminf(fmaxf(x, -1.0f), 1.0f); }

// 4 f32 -> v4h (clang emits 2x v_cvt_pk_f16_f32, stored as one b64)
__device__ __forceinline__ v4h cvt4(float a, float b, float c, float d) {
  v4h r;
  r[0] = (_Float16)a; r[1] = (_Float16)b; r[2] = (_Float16)c; r[3] = (_Float16)d;
  return r;
}

// CDNA5 wave32 WMMA fragment maps (CDNA5 ISA §7.12.2):
//  A 16x32 f16 : lane holds row m = L%16; element e -> k = (e<8 ? e : e+8) + (L>=16 ? 8 : 0)
//  B 32x16 f16 : lane holds col n = L%16; element e -> k = (e<8 ? e : e+8) + (L>=16 ? 8 : 0)
//  C/D 16x16 f32: lane holds col n = L%16; vgpr j -> row m = j + (L>=16 ? 8 : 0)
// A-element pairs (2p,2p+1) are (k,k+1) with k even; window col k%6 is even -> the two
// taps are contiguous & 4B-aligned in the input row: one ds_load_b32 per fragment dword.
// All "invalid" fragment elements keep in-bounds addresses: loads are unconditional and
// the zero-fill is a v_cndmask, never an exec-mask branch.

__global__ __launch_bounds__(128) void lila_fused(
    const float* __restrict__ inp,
    const float* __restrict__ W1, const float* __restrict__ b1,
    const float* __restrict__ W2, const float* __restrict__ b2,
    const float* __restrict__ W3, const float* __restrict__ b3,
    const float* __restrict__ V1, const float* __restrict__ c1,
    const float* __restrict__ V2, const float* __restrict__ c2,
    const float* __restrict__ V3, const float* __restrict__ c3,
    float* __restrict__ out)
{
  __shared__ _Float16 s_in  [16][392];       // shared input slab (4m%64 bank stagger)
  __shared__ _Float16 s_hT  [4][2][16][20];  // per-wave transposed h tiles [k][m]
  __shared__ _Float16 s_visT[144][20];       // transposed vision [k][m] (disjoint cols/wave)
  __shared__ _Float16 s_eT  [2][32][20];     // transposed e1/e2 tiles [k][m]

  const int  tid    = threadIdx.x;
  const int  wid    = tid >> 5;          // wave 0..3
  const int  lane   = tid & 31;
  const int  n      = lane & 15;
  const bool hiHalf = lane >= 16;
  const int  aK     = hiHalf ? 8  : 0;   // A-frag per-lane k offset
  const int  bK     = hiHalf ? 16 : 0;   // unused (B frags use aK)
  const int  mRow   = n;                 // A-frag row owned by this lane
  const int  mC     = hiHalf ? 8 : 0;    // C-frag row base
  const long r0     = (long)blockIdx.x * 16;

  // ---- stage 16-row slab (contiguous, 16B-aligned) via b128 loads --------
  {
    const float4* g = (const float4*)(inp + r0 * 385);
    for (int i = tid; i < 1540; i += 128) {    // 16*385/4
      float4 v = g[i];
      int f = i * 4;
      int r = f / 385;
      int c = f - r * 385;
      if (c <= 381) {                          // fast path: one base, imm offsets
        _Float16* p = &s_in[r][c];
        p[0] = (_Float16)v.x; p[1] = (_Float16)v.y;
        p[2] = (_Float16)v.z; p[3] = (_Float16)v.w;
      } else {                                 // rare row-crossing float4
#pragma unroll
        for (int t = 0; t < 4; ++t) {
          int rr = r, cc = c + t;
          if (cc >= 385) { rr += 1; cc -= 385; }
          float x = (t == 0) ? v.x : (t == 1) ? v.y : (t == 2) ? v.z : v.w;
          s_in[rr][cc] = (_Float16)x;
        }
      }
    }
  }

  // ---- pre-swizzled B-operand weight fragments (clamped addr + cndmask) --
  auto loadB = [&](const float* W, int rows, int cols, int kbase, int ncol) {
    v16h b;
    const int ncc = (ncol < cols) ? ncol : 0;
#pragma unroll
    for (int e = 0; e < 16; ++e) {
      int k  = kbase + ((e < 8) ? e : e + 8) + aK;
      int kc = (k < rows) ? k : 0;
      float v = W[kc * cols + ncc];            // unconditional, in-bounds
      b[e] = (k < rows && ncol < cols) ? (_Float16)v : (_Float16)0.0f;
    }
    return b;
  };

  const v16h w1b0 = loadB(W1, 54, 12, 0,  n);
  const v16h w1b1 = loadB(W1, 54, 12, 32, n);
  const v16h w2b  = loadB(W2, 12, 12, 0,  n);
  const v16h w3b  = loadB(W3, 12, 4,  0,  n);
  const float bb1 = (n < 12) ? b1[n < 12 ? n : 0] : 0.0f;
  const float bb2 = (n < 12) ? b2[n < 12 ? n : 0] : 0.0f;
  const float bb3 = (n < 4)  ? b3[n < 4  ? n : 0] : 0.0f;

  // ---- per-lane pair offsets (in dwords) for the window gather, hoisted --
  // pair p -> k0 = (p<4 ? 2p : 2p+8) + aK ; second chunk k1 = k0+32
  // valid(k1): lo lanes p<7, hi lanes p<4 ; invalid pairs use a safe dummy addr
  int offH0[8], offH1[8];
#pragma unroll
  for (int p = 0; p < 8; ++p) {
    int k0 = ((p < 4) ? 2 * p : 2 * p + 8) + aK;
    { int g0 = k0 / 6, j = k0 - 6 * g0;
      offH0[p] = (((g0 / 3) * 48 + (g0 % 3) * 6 + j) >> 1); }
    int k1 = k0 + 32;
    int kk = (k1 < 54) ? k1 : 0;
    { int g1 = kk / 6, j = kk - 6 * g1;
      offH1[p] = (((g1 / 3) * 48 + (g1 % 3) * 6 + j) >> 1); }
  }
  const unsigned* rowIn = (const unsigned*)&s_in[mRow][0];

  __syncthreads();                              // slab visible to all waves

  // ---- stage A: 9 windows per wave, 54 -> 12 -> 12 -> 4 ------------------
  for (int w = wid * 9; w < wid * 9 + 9; ++w) {
    const int wx = w / 6, wy = w % 6;
    const int baseH = (wx * 48 + wy * 6) >> 1;  // wave-uniform

    v8u ua0, ua1;
#pragma unroll
    for (int p = 0; p < 8; ++p) {
      ua0[p] = rowIn[baseH + offH0[p]];         // always valid
      if (p < 4) {
        ua1[p] = rowIn[baseH + offH1[p]];       // valid for both halves
      } else if (p < 7) {
        unsigned v = rowIn[baseH + offH1[p]];   // unconditional load
        ua1[p] = hiHalf ? 0u : v;               // v_cndmask
      } else {
        ua1[p] = 0u;                            // never valid
      }
    }
    v16h a0 = __builtin_bit_cast(v16h, ua0);
    v16h a1 = __builtin_bit_cast(v16h, ua1);

    v8f h1 = {};
    h1 = WMMA(a0, w1b0, h1);
    h1 = WMMA(a1, w1b1, h1);

    if (n < 12) {                               // spill column n as 2x b64
      *(v4h*)&s_hT[wid][0][n][mC] =
          cvt4(crelu1(h1[0] + bb1), crelu1(h1[1] + bb1),
               crelu1(h1[2] + bb1), crelu1(h1[3] + bb1));
      *(v4h*)&s_hT[wid][0][n][mC + 4] =
          cvt4(crelu1(h1[4] + bb1), crelu1(h1[5] + bb1),
               crelu1(h1[6] + bb1), crelu1(h1[7] + bb1));
    }
    WAVE_FENCE();                               // wave-private tile: no barrier

    v16h ah;                                    // K = 12, padded to 32
#pragma unroll
    for (int e = 0; e < 8; ++e) {               // k = e + aK (<= 15, in-bounds)
      _Float16 v = s_hT[wid][0][e + aK][mRow];
      ah[e] = (e < 4 || !hiHalf) ? v : (_Float16)0.0f;
    }
#pragma unroll
    for (int e = 8; e < 16; ++e) ah[e] = (_Float16)0.0f;

    v8f h2 = {};
    h2 = WMMA(ah, w2b, h2);

    if (n < 12) {
      *(v4h*)&s_hT[wid][1][n][mC] =
          cvt4(crelu1(h2[0] + bb2), crelu1(h2[1] + bb2),
               crelu1(h2[2] + bb2), crelu1(h2[3] + bb2));
      *(v4h*)&s_hT[wid][1][n][mC + 4] =
          cvt4(crelu1(h2[4] + bb2), crelu1(h2[5] + bb2),
               crelu1(h2[6] + bb2), crelu1(h2[7] + bb2));
    }
    WAVE_FENCE();

    v16h ah2;
#pragma unroll
    for (int e = 0; e < 8; ++e) {
      _Float16 v = s_hT[wid][1][e + aK][mRow];
      ah2[e] = (e < 4 || !hiHalf) ? v : (_Float16)0.0f;
    }
#pragma unroll
    for (int e = 8; e < 16; ++e) ah2[e] = (_Float16)0.0f;

    v8f sg = {};
    sg = WMMA(ah2, w3b, sg);

    if (n < 4) {                                // sight -> transposed vision
      const int col = w * 4 + n;                // disjoint column range per wave
      *(v4h*)&s_visT[col][mC] =
          cvt4(crelu1(sg[0] + bb3), crelu1(sg[1] + bb3),
               crelu1(sg[2] + bb3), crelu1(sg[3] + bb3));
      *(v4h*)&s_visT[col][mC + 4] =
          cvt4(crelu1(sg[4] + bb3), crelu1(sg[5] + bb3),
               crelu1(sg[6] + bb3), crelu1(sg[7] + bb3));
    }
  }
  __syncthreads();                              // all 36 windows complete

  // ---- stage B: vision(145, pad 160) -> 32 -> 32 -> 1 --------------------
  // waves 0/1 each own one 16-wide N-tile; waves 2/3 just keep the barriers.
  if (wid < 2) {
    v16h va[5];
#pragma unroll
    for (int cb = 0; cb < 5; ++cb) {
#pragma unroll
      for (int e = 0; e < 16; ++e) {
        int kE = (e < 8) ? e : e + 8;
        int k  = cb * 32 + kE + aK;
        if (cb < 4 || e < 8) {                  // k <= 143: plain load
          va[cb][e] = s_visT[k][mRow];
        } else if (e == 8) {                    // k == 144 on lo half only
          _Float16 t = s_in[mRow][384];
          va[cb][e] = hiHalf ? (_Float16)0.0f : t;
        } else {
          va[cb][e] = (_Float16)0.0f;           // 145..159 zero pad
        }
      }
    }

    const int nc = n + 16 * wid;                // e1 = crelu(vision @ V1 + c1)
    v8f acc = {};
#pragma unroll
    for (int cb = 0; cb < 5; ++cb) {
      v16h vb = loadB(V1, 145, 32, cb * 32, nc);
      acc = WMMA(va[cb], vb, acc);
    }
    const float bc = c1[nc];
    *(v4h*)&s_eT[0][nc][mC] =
        cvt4(crelu1(acc[0] + bc), crelu1(acc[1] + bc),
             crelu1(acc[2] + bc), crelu1(acc[3] + bc));
    *(v4h*)&s_eT[0][nc][mC + 4] =
        cvt4(crelu1(acc[4] + bc), crelu1(acc[5] + bc),
             crelu1(acc[6] + bc), crelu1(acc[7] + bc));
  }
  __syncthreads();

  if (wid < 2) {
    v16h ae;                                    // K = 32, fully valid
#pragma unroll
    for (int e = 0; e < 16; ++e) {
      int k = ((e < 8) ? e : e + 8) + aK;
      ae[e] = s_eT[0][k][mRow];
    }
    const int nc = n + 16 * wid;                // e2 = crelu(e1 @ V2 + c2)
    v16h vb = loadB(V2, 32, 32, 0, nc);
    v8f acc = {};
    acc = WMMA(ae, vb, acc);
    const float bc = c2[nc];
    *(v4h*)&s_eT[1][nc][mC] =
        cvt4(crelu1(acc[0] + bc), crelu1(acc[1] + bc),
             crelu1(acc[2] + bc), crelu1(acc[3] + bc));
    *(v4h*)&s_eT[1][nc][mC + 4] =
        cvt4(crelu1(acc[4] + bc), crelu1(acc[5] + bc),
             crelu1(acc[6] + bc), crelu1(acc[7] + bc));
  }
  __syncthreads();

  if (wid == 0) {
    v16h af;
#pragma unroll
    for (int e = 0; e < 16; ++e) {
      int k = ((e < 8) ? e : e + 8) + aK;
      af[e] = s_eT[1][k][mRow];
    }
    v16h vb3 = loadB(V3, 32, 1, 0, n);          // out = crelu(e2 @ V3 + c3)
    v8f acc = {};
    acc = WMMA(af, vb3, acc);
    if (n == 0) {
      const float bc = c3[0];
      float4 o0 = make_float4(crelu1(acc[0] + bc), crelu1(acc[1] + bc),
                              crelu1(acc[2] + bc), crelu1(acc[3] + bc));
      float4 o1 = make_float4(crelu1(acc[4] + bc), crelu1(acc[5] + bc),
                              crelu1(acc[6] + bc), crelu1(acc[7] + bc));
      *(float4*)(out + r0 + mC)     = o0;       // 16B aligned (r0+mC % 8 == 0)
      *(float4*)(out + r0 + mC + 4) = o1;
    }
  }
}

extern "C" void kernel_launch(void* const* d_in, const int* in_sizes, int n_in,
                              void* d_out, int out_size, void* d_ws, size_t ws_size,
                              hipStream_t stream) {
  const float* inp = (const float*)d_in[0];
  const float* W1  = (const float*)d_in[1];
  const float* b1  = (const float*)d_in[2];
  const float* W2  = (const float*)d_in[3];
  const float* b2  = (const float*)d_in[4];
  const float* W3  = (const float*)d_in[5];
  const float* b3  = (const float*)d_in[6];
  const float* V1  = (const float*)d_in[7];
  const float* c1  = (const float*)d_in[8];
  const float* V2  = (const float*)d_in[9];
  const float* c2  = (const float*)d_in[10];
  const float* V3  = (const float*)d_in[11];
  const float* c3  = (const float*)d_in[12];

  const int B    = in_sizes[0] / 385;   // 131072
  const int grid = B / 16;              // 16-row tile per block, 4 waves cooperating
  lila_fused<<<grid, 128, 0, stream>>>(inp, W1, b1, W2, b2, W3, b3,
                                       V1, c1, V2, c2, V3, c3, (float*)d_out);
}
